// HomogeneousHORNNetwork_5652176962305
// MI455X (gfx1250) — matrix operand order repaired and checked by reading, not compiled, not run on hardware
//
#include <hip/hip_runtime.h>
#include <math.h>

#define NUM_NODES   94
#define NPAD        96
#define SEQ_LEN     784
#define OUT_CLASSES 10
#define BTILE       16      // batch rows per workgroup (one WMMA M-tile)
#define THREADS     96      // 3 waves; wave w owns N-tiles {2w, 2w+1}

typedef __attribute__((ext_vector_type(16))) __bf16       v16bf;
typedef __attribute__((ext_vector_type(8)))  float        v8f;
typedef __attribute__((ext_vector_type(4)))  unsigned int v4u;

union BF16x16 {                 // one WMMA bf16 operand fragment (8 VGPRs)
    unsigned short u[16];
    v4u            q[2];
    v16bf          v;
};

__device__ __forceinline__ unsigned short f2bf(float f) {
    unsigned int u = __float_as_uint(f);
    u += 0x7FFFu + ((u >> 16) & 1u);       // round-to-nearest-even
    return (unsigned short)(u >> 16);
}

// Branch-free tanh on the serial critical path.
// Prefer native v_tanh_f32 (gfx1250 TRANS op) when the toolchain exposes it;
// otherwise tanh(a) = sign(a) * (1 - 2/(1 + exp2(2*log2e*|a|))) built from
// v_exp_f32 + v_rcp_f32 only (no EXEC divergence; exact +-1 saturation via inf).
__device__ __forceinline__ float fast_tanh(float a) {
#if __has_builtin(__builtin_amdgcn_tanhf)
    return __builtin_amdgcn_tanhf(a);
#else
    const float K = 2.8853900817779268f;           // 2 * log2(e)
    float ax = __builtin_fabsf(a);
    float e  = __builtin_amdgcn_exp2f(K * ax);     // v_exp_f32
    float r  = __builtin_amdgcn_rcpf(1.0f + e);    // v_rcp_f32
    float m  = __builtin_fmaf(-2.0f, r, 1.0f);     // 1 - 2/(1+e)  in [0,1)
    return __builtin_copysignf(m, a);
#endif
}

__global__ __launch_bounds__(THREADS) void horn_scan_kernel(
    const float* __restrict__ in_seq,   // (B, 784)
    const float* __restrict__ w_ih,     // (94, 1)
    const float* __restrict__ b_ih,     // (94)
    const float* __restrict__ w_hh,     // (94, 94)
    const float* __restrict__ b_hh,     // (94)
    const float* __restrict__ w_ro,     // (10, 94)
    const float* __restrict__ b_ro,     // (10)
    float* __restrict__ out)            // (B, 10)
{
    __shared__ __align__(16) float          in_sh[BTILE][SEQ_LEN];   // 50 KB input tile
    __shared__ __align__(16) unsigned short y_sh[2][BTILE][NPAD];    // double-buffered y (bf16)

    const int tid   = threadIdx.x;
    const int wave  = tid >> 5;          // 0..2
    const int lane  = tid & 31;
    const int nl    = lane & 15;
    const int kh    = lane >> 4;         // lane half selects K sub-range / M sub-range
    const int bbase = blockIdx.x * BTILE;

    // ---- stage this block's 16x784 input tile into LDS (one-time cost) ----
    for (int idx = tid; idx < BTILE * SEQ_LEN; idx += THREADS) {
        int m = idx / SEQ_LEN, t = idx - m * SEQ_LEN;
        in_sh[m][t] = in_seq[(bbase + m) * SEQ_LEN + t];
    }
    // ---- zero both y buffers (state starts at 0; pad columns stay 0) ----
    for (int idx = tid; idx < 2 * BTILE * NPAD; idx += THREADS)
        ((unsigned short*)y_sh)[idx] = 0;

    const float ISN   = 1.0f / sqrtf((float)NUM_NODES);
    const float OMEGA = 2.0f * (float)M_PI / 28.0f;
    const float OM2   = OMEGA * OMEGA;

    // ---- build resident B fragments of w_hh^T (bf16) + per-node constants ----
    // B layout (32x16 bf16): lane holds column n = nl; element e<8 -> k = kbase + 8*kh + e, e>=8 -> k = kbase + 16 + 8*kh + (e-8)
    BF16x16 Bf[2][3];
    float wiISN[2], bISN[2];
    int nbase[2];
#pragma unroll
    for (int j = 0; j < 2; ++j) {
        int n = (2 * wave + j) * 16 + nl;
        nbase[j] = n;
        bool nok = (n < NUM_NODES);
        wiISN[j] = (nok ? w_ih[n] : 0.0f) * ISN;
        bISN[j]  = ((nok ? b_ih[n] : 0.0f) + (nok ? b_hh[n] : 0.0f)) * ISN;
#pragma unroll
        for (int c = 0; c < 3; ++c)
#pragma unroll
            for (int e = 0; e < 16; ++e) {
                int k = 32 * c + 8 * kh + (e < 8 ? e : e + 8);
                float w = (nok && k < NUM_NODES) ? w_hh[n * NUM_NODES + k] : 0.0f;
                Bf[j][c].u[e] = f2bf(w);
            }
    }

    // ---- oscillator state in WMMA C-layout registers: elem r -> (m = r + 8*kh, n = nbase[j]) ----
    v8f x[2], y[2];
#pragma unroll
    for (int j = 0; j < 2; ++j)
#pragma unroll
        for (int r = 0; r < 8; ++r) { x[j][r] = 0.0f; y[j][r] = 0.0f; }

    __syncthreads();

    // ================= 784 serial steps, fully on-chip =================
    for (int t = 0; t < SEQ_LEN; ++t) {
        const int p = t & 1;

        // A fragments of y (16x32 bf16): lane row m = nl;
        // elems 0..7 -> k = kbase + 8*kh + e ; elems 8..15 -> k = kbase + 16 + 8*kh + (e-8)
        const v4u* yrow = (const v4u*)(&y_sh[p][nl][0]);   // 12 x 16B per row
        BF16x16 A[3];
#pragma unroll
        for (int c = 0; c < 3; ++c) {
            A[c].q[0] = yrow[4 * c + kh];
            A[c].q[1] = yrow[4 * c + 2 + kh];
        }

        // drive signal for my 8 batch rows (m = r + 8*kh)
        float sv[8];
#pragma unroll
        for (int r = 0; r < 8; ++r) sv[r] = in_sh[8 * kh + r][t];

#pragma unroll
        for (int j = 0; j < 2; ++j) {
            v8f acc = {0.f, 0.f, 0.f, 0.f, 0.f, 0.f, 0.f, 0.f};
            acc = __builtin_amdgcn_wmma_f32_16x16x32_bf16(false, A[0].v, false, Bf[j][0].v,
                                                          (short)0, acc, false, false);
            acc = __builtin_amdgcn_wmma_f32_16x16x32_bf16(false, A[1].v, false, Bf[j][1].v,
                                                          (short)0, acc, false, false);
            acc = __builtin_amdgcn_wmma_f32_16x16x32_bf16(false, A[2].v, false, Bf[j][2].v,
                                                          (short)0, acc, false, false);
#pragma unroll
            for (int r = 0; r < 8; ++r) {
                // A = s*w_ih*ISN + acc*ISN + (b_ih+b_hh)*ISN  (2 FMAs)
                float Aval  = __builtin_fmaf(sv[r], wiISN[j],
                              __builtin_fmaf(acc[r], ISN, bISN[j]));
                float accel = __builtin_fmaf(0.5f, fast_tanh(Aval),
                              __builtin_fmaf(-0.2f, y[j][r], -OM2 * x[j][r]));
                float xn = x[j][r] + y[j][r];
                float yn = y[j][r] + accel;
                x[j][r] = xn;
                y[j][r] = yn;
                y_sh[1 - p][8 * kh + r][nbase[j]] = f2bf(yn);
            }
        }
        __syncthreads();   // double buffering -> single barrier per step
    }

    // ================= readout: out = x_final @ w_ro^T + b_ro =================
    float* x_sh = &in_sh[0][0];   // reuse input LDS (16 x NPAD fp32)
#pragma unroll
    for (int j = 0; j < 2; ++j)
#pragma unroll
        for (int r = 0; r < 8; ++r)
            x_sh[(8 * kh + r) * NPAD + nbase[j]] = x[j][r];
    __syncthreads();

    for (int idx = tid; idx < BTILE * OUT_CLASSES; idx += THREADS) {
        int b = idx / OUT_CLASSES, o = idx - b * OUT_CLASSES;
        float s = b_ro[o];
        for (int n = 0; n < NUM_NODES; ++n)
            s += x_sh[b * NPAD + n] * w_ro[o * NUM_NODES + n];
        out[(bbase + b) * OUT_CLASSES + o] = s;
    }
}

extern "C" void kernel_launch(void* const* d_in, const int* in_sizes, int n_in,
                              void* d_out, int out_size, void* d_ws, size_t ws_size,
                              hipStream_t stream) {
    (void)n_in; (void)out_size; (void)d_ws; (void)ws_size;
    const float* in_seq = (const float*)d_in[0];
    const float* w_ih   = (const float*)d_in[1];
    const float* b_ih   = (const float*)d_in[2];
    const float* w_hh   = (const float*)d_in[3];
    const float* b_hh   = (const float*)d_in[4];
    const float* w_ro   = (const float*)d_in[5];
    const float* b_ro   = (const float*)d_in[6];
    float* out = (float*)d_out;

    const int batch = in_sizes[0] / SEQ_LEN;   // 1024
    const int grid  = batch / BTILE;           // 64 workgroups, 3 waves each

    horn_scan_kernel<<<grid, THREADS, 0, stream>>>(in_seq, w_ih, b_ih, w_hh, b_hh,
                                                   w_ro, b_ro, out);
}
